// GCN2_16329465659966
// MI455X (gfx1250) — matrix operand order repaired and checked by reading, not compiled, not run on hardware
//
#include <hip/hip_runtime.h>
#include <hip/hip_bf16.h>

#define DH 128          // feature dim (D == H == 128)
#define EPSLN 1e-5f

typedef __attribute__((ext_vector_type(16))) __bf16 v16bf;
typedef __attribute__((ext_vector_type(8)))  __bf16 v8bf;
typedef __attribute__((ext_vector_type(8)))  float  v8f;

// Hardware f32 atomic add (no-return form -> STOREcnt, executed at L2).
// Guarantees global_atomic_add_f32 instead of a possible CAS-loop expansion.
__device__ __forceinline__ void atomic_add_f32_hw(float* p, float v) {
    asm volatile("global_atomic_add_f32 %0, %1, off"
                 :: "v"(reinterpret_cast<unsigned long long>(p)), "v"(v)
                 : "memory");
}

// ---------------------------------------------------------------- utilities
__global__ __launch_bounds__(256) void gcn_zero_f32(float* p, long long n) {
    long long i = (long long)blockIdx.x * blockDim.x + threadIdx.x;
    if (i < n) p[i] = 0.0f;
}

// deg[dst] += 1 per edge
__global__ __launch_bounds__(256) void gcn_deg(const int* dst, float* deg, int E) {
    int e = blockIdx.x * blockDim.x + threadIdx.x;
    if (e < E) atomic_add_f32_hw(&deg[dst[e]], 1.0f);
}

// dinv = rsqrt(deg + 1)  (A + I degree), in place
__global__ __launch_bounds__(256) void gcn_dinv(float* deg, int n) {
    int i = blockIdx.x * blockDim.x + threadIdx.x;
    if (i < n) deg[i] = rsqrtf(deg[i] + 1.0f);
}

// f32 -> bf16 elementwise
__global__ __launch_bounds__(256) void gcn_cvt_bf16(const float* x, __bf16* y, long long n) {
    long long i = (long long)blockIdx.x * blockDim.x + threadIdx.x;
    if (i < n) y[i] = (__bf16)x[i];
}

// Wt[n][k] = (bf16) W[k][n]   (128x128)
__global__ __launch_bounds__(256) void gcn_wtrans(const float* W, __bf16* Wt) {
    int i = blockIdx.x * blockDim.x + threadIdx.x;   // 0 .. 16383
    int k = i >> 7, n = i & 127;
    Wt[n * DH + k] = (__bf16)W[k * DH + n];
}

// ------------------------------------------------- WMMA GEMM: C = A @ W
// A: [n, 128] bf16 row-major.  Wt: [128,128] bf16 with Wt[col][k] = W[k][col].
// One wave -> one 16x16 tile of C; 8 waves/block cover the 128-wide N dim.
__global__ __launch_bounds__(256)
void gcn_gemm_bf16(const __bf16* __restrict__ A, const __bf16* __restrict__ Wt,
                   float* __restrict__ C, int n) {
    const int lane = threadIdx.x & 31;
    const int wave = threadIdx.x >> 5;        // N-tile index (0..7)
    const int m0   = blockIdx.x * 16;
    const int n0   = wave * 16;
    const int half = lane >> 4;               // 0: lanes 0-15, 1: lanes 16-31
    const int l    = lane & 15;

    // A fragment rows: lane l / l+16 both hold row m0+l (different K halves)
    int arow = m0 + l; if (arow >= n) arow = n - 1;          // clamp (uniform-safe)
    const __bf16* Ap = A  + (long long)arow * DH;
    const __bf16* Bp = Wt + (long long)(n0 + l) * DH;        // column n0+l of W

    v8f acc = {};
#pragma unroll
    for (int k0 = 0; k0 < DH; k0 += 32) {
        v16bf a, b;
        // 16-bit A 16x32 layout: e0..7 -> K=k0+8*half..+7 ; e8..15 -> K=k0+16+8*half..+7
        *(v8bf*)&a         = *(const v8bf*)(Ap + k0 + half * 8);
        *(((v8bf*)&a) + 1) = *(const v8bf*)(Ap + k0 + 16 + half * 8);
        // 16-bit B 32x16 layout: lane's column, e0..7 -> K=k0+8*half..+7 ; e8..15 -> K=k0+16+8*half..+7
        *(v8bf*)&b         = *(const v8bf*)(Bp + k0 + half * 8);
        *(((v8bf*)&b) + 1) = *(const v8bf*)(Bp + k0 + 16 + half * 8);
        acc = __builtin_amdgcn_wmma_f32_16x16x32_bf16(false, a, false, b,
                                                      (short)0, acc, false, false);
    }
    // C/D layout: VGPR r -> (M = m0 + 8*half + r, N = n0 + l)
    float* Cp = C + (long long)(m0 + half * 8) * DH + (n0 + l);
    if (m0 + 16 <= n) {                    // uniform fast path: whole tile in range
#pragma unroll
        for (int r = 0; r < 8; ++r) Cp[(long long)r * DH] = acc[r];
    } else {                               // edge tile: guarded stores
#pragma unroll
        for (int r = 0; r < 8; ++r)
            if (m0 + half * 8 + r < n) Cp[(long long)r * DH] = acc[r];
    }
}

// ------------------------------------------ edge scatter: agg[dst] += xw[src]*norm
// one wave per edge; each lane moves a float4 (512B coalesced per edge)
__global__ __launch_bounds__(256)
void gcn_scatter(const int* __restrict__ src, const int* __restrict__ dst,
                 const float* __restrict__ dinv, const float* __restrict__ xw,
                 float* __restrict__ agg, int E) {
    int lane = threadIdx.x & 31;
    int e = blockIdx.x * 8 + (threadIdx.x >> 5);
    if (e >= E) return;
    int s = src[e], d = dst[e];
    float norm = dinv[s] * dinv[d];
    const float4 v = *(const float4*)(xw + (long long)s * DH + lane * 4);
    float* base = agg + (long long)d * DH + lane * 4;
    atomic_add_f32_hw(base + 0, v.x * norm);
    atomic_add_f32_hw(base + 1, v.y * norm);
    atomic_add_f32_hw(base + 2, v.z * norm);
    atomic_add_f32_hw(base + 3, v.w * norm);
}

// h = relu(agg + xw*dinv^2 + b)  -> bf16 (feeds next WMMA GEMM)
__global__ __launch_bounds__(256)
void gcn_combine_relu(const float* __restrict__ agg, const float* __restrict__ xw,
                      const float* __restrict__ dinv, const float* __restrict__ b,
                      __bf16* __restrict__ h, long long total) {
    long long i = (long long)blockIdx.x * blockDim.x + threadIdx.x;
    if (i >= total) return;
    int node = (int)(i >> 7), j = (int)(i & 127);
    float di = dinv[node];
    float v = agg[i] + xw[i] * di * di + b[j];
    h[i] = (__bf16)fmaxf(v, 0.0f);
}

// layer-2 epilogue fused with graph mean-pool numerator
__global__ __launch_bounds__(256)
void gcn_combine_pool(const float* __restrict__ agg, const float* __restrict__ xw,
                      const float* __restrict__ dinv, const float* __restrict__ b,
                      const int* __restrict__ batch, float* __restrict__ sums,
                      long long total) {
    long long i = (long long)blockIdx.x * blockDim.x + threadIdx.x;
    if (i >= total) return;
    int node = (int)(i >> 7), j = (int)(i & 127);
    float di = dinv[node];
    float v = agg[i] + xw[i] * di * di + b[j];
    atomic_add_f32_hw(&sums[(long long)batch[node] * DH + j], v);
}

__global__ __launch_bounds__(256) void gcn_counts(const int* batch, float* cnt, int n) {
    int i = blockIdx.x * blockDim.x + threadIdx.x;
    if (i < n) atomic_add_f32_hw(&cnt[batch[i]], 1.0f);
}

// ------------------------- per-graph: mean -> GRU(h0=0) -> relu -> LN -> linear
__global__ __launch_bounds__(128)
void gcn_head(const float* __restrict__ sums, const float* __restrict__ cnt,
              const float* __restrict__ W_ih, const float* __restrict__ b_ih,
              const float* __restrict__ b_hh, const float* __restrict__ W_lin,
              const float* __restrict__ b_lin, float* __restrict__ out) {
    __shared__ float gsh[DH];
    __shared__ float red[DH];
    int g = blockIdx.x, j = threadIdx.x;
    float c = fmaxf(cnt[g], 1.0f);
    float gv = sums[(long long)g * DH + j] / c;
    gsh[j] = gv;
    __syncthreads();

    float ir = b_ih[j], iz = b_ih[DH + j], in_ = b_ih[2 * DH + j];
    const float* wr = W_ih + (long long)j * DH;
    const float* wz = W_ih + (long long)(DH + j) * DH;
    const float* wn = W_ih + (long long)(2 * DH + j) * DH;
#pragma unroll 4
    for (int k = 0; k < DH; ++k) {
        float gk = gsh[k];
        ir = fmaf(gk, wr[k], ir);
        iz = fmaf(gk, wz[k], iz);
        in_ = fmaf(gk, wn[k], in_);
    }
    // gh = b_hh since h0 == 0
    float r = 1.0f / (1.0f + expf(-(ir + b_hh[j])));
    float z = 1.0f / (1.0f + expf(-(iz + b_hh[DH + j])));
    float nn = tanhf(in_ + r * b_hh[2 * DH + j]);
    float h = fmaxf((1.0f - z) * nn, 0.0f);     // + z*h0 (=0), then relu

    // layer-norm over the 128 features of this graph
    red[j] = h; __syncthreads();
    for (int s = 64; s > 0; s >>= 1) { if (j < s) red[j] += red[j + s]; __syncthreads(); }
    float mu = red[0] * (1.0f / DH); __syncthreads();
    float d = h - mu;
    red[j] = d * d; __syncthreads();
    for (int s = 64; s > 0; s >>= 1) { if (j < s) red[j] += red[j + s]; __syncthreads(); }
    float var = red[0] * (1.0f / DH); __syncthreads();
    float nv = d * rsqrtf(var + EPSLN);
    red[j] = nv * W_lin[j]; __syncthreads();
    for (int s = 64; s > 0; s >>= 1) { if (j < s) red[j] += red[j + s]; __syncthreads(); }
    if (j == 0) out[g] = red[0] + b_lin[0];
}

// ---------------------------------------------------------------- launcher
extern "C" void kernel_launch(void* const* d_in, const int* in_sizes, int n_in,
                              void* d_out, int out_size, void* d_ws, size_t ws_size,
                              hipStream_t stream) {
    const float* x     = (const float*)d_in[0];
    const int*   ei    = (const int*)d_in[1];
    const int*   batch = (const int*)d_in[2];
    const float* W1    = (const float*)d_in[3];
    const float* b1    = (const float*)d_in[4];
    const float* W2    = (const float*)d_in[5];
    const float* b2    = (const float*)d_in[6];
    const float* W_ih  = (const float*)d_in[7];
    /* W_hh (d_in[8]) unused: h0 == 0 */
    const float* b_ih  = (const float*)d_in[9];
    const float* b_hh  = (const float*)d_in[10];
    const float* W_lin = (const float*)d_in[11];
    const float* b_lin = (const float*)d_in[12];
    float* out = (float*)d_out;

    const int n = in_sizes[0] / DH;        // 100000 nodes
    const int E = in_sizes[1] / 2;         // 1.6M edges
    const int G = out_size;                // 2048 graphs
    const int* src = ei;
    const int* dst = ei + E;
    const long long nd = (long long)n * DH;

    // workspace carve-out (all offsets 16B aligned for n multiple of 4)
    char* ws = (char*)d_ws;
    float*  dinv = (float*)ws;                    ws += (size_t)((n + 63) & ~63) * 4;
    float*  xw   = (float*)ws;                    ws += (size_t)nd * 4;
    float*  agg  = (float*)ws;                    ws += (size_t)nd * 4;
    __bf16* hbf  = (__bf16*)ws;                   ws += (size_t)nd * 2;
    __bf16* Wt1  = (__bf16*)ws;                   ws += (size_t)DH * DH * 2;
    __bf16* Wt2  = (__bf16*)ws;                   ws += (size_t)DH * DH * 2;
    float*  sums = (float*)ws;                    ws += (size_t)G * DH * 4;
    float*  cnt  = (float*)ws;

    const int B = 256;
    const int gN    = (n + B - 1) / B;
    const int gE    = (E + B - 1) / B;
    const int gND   = (int)((nd + B - 1) / B);
    const int gEdge = (E + 7) / 8;                // 1 wave per edge
    const int gTile = (n + 15) / 16;              // WMMA M-tiles

    // degrees / dinv
    gcn_zero_f32<<<gN, B, 0, stream>>>(dinv, n);
    gcn_deg<<<gE, B, 0, stream>>>(dst, dinv, E);
    gcn_dinv<<<gN, B, 0, stream>>>(dinv, n);

    // bf16 operands
    gcn_cvt_bf16<<<gND, B, 0, stream>>>(x, hbf, nd);
    gcn_wtrans<<<64, B, 0, stream>>>(W1, Wt1);
    gcn_wtrans<<<64, B, 0, stream>>>(W2, Wt2);

    // ---- layer 1: xw = x@W1 ; agg = scatter ; h1 = relu(...) -> bf16
    gcn_gemm_bf16<<<gTile, B, 0, stream>>>(hbf, Wt1, xw, n);
    gcn_zero_f32<<<gND, B, 0, stream>>>(agg, nd);
    gcn_scatter<<<gEdge, B, 0, stream>>>(src, dst, dinv, xw, agg, E);
    gcn_combine_relu<<<gND, B, 0, stream>>>(agg, xw, dinv, b1, hbf, nd);

    // ---- layer 2: xw = h1@W2 ; agg = scatter ; fused epilogue + mean-pool sums
    gcn_gemm_bf16<<<gTile, B, 0, stream>>>(hbf, Wt2, xw, n);
    gcn_zero_f32<<<gND, B, 0, stream>>>(agg, nd);
    gcn_scatter<<<gEdge, B, 0, stream>>>(src, dst, dinv, xw, agg, E);

    gcn_zero_f32<<<(G * DH + B - 1) / B, B, 0, stream>>>(sums, (long long)G * DH);
    gcn_zero_f32<<<(G + B - 1) / B, B, 0, stream>>>(cnt, G);
    gcn_combine_pool<<<gND, B, 0, stream>>>(agg, xw, dinv, b2, batch, sums, nd);
    gcn_counts<<<gN, B, 0, stream>>>(batch, cnt, n);

    // ---- per-graph head: mean -> GRU -> relu -> LN -> linear
    gcn_head<<<G, DH, 0, stream>>>(sums, cnt, W_ih, b_ih, b_hh, W_lin, b_lin, out);
}
